// LSTM_53498112639105
// MI455X (gfx1250) — matrix operand order repaired and checked by reading, not compiled, not run on hardware
//
#include <hip/hip_runtime.h>

typedef __attribute__((ext_vector_type(16))) _Float16 v16h;
typedef __attribute__((ext_vector_type(8)))  _Float16 v8h;
typedef __attribute__((ext_vector_type(8)))  float    v8f;

#define B_SZ   4096
#define T_SZ   128
#define D_SZ   32
#define H_SZ   50
#define FC_SZ  25
#define HP     64     // hidden padded to 64
#define GP     256    // gates padded: 4*HP
#define WAVES  2
#define BTILE  16     // batch rows per wave (== WMMA M)

#define LOG2E 1.4426950408889634f

// ---- branch-free activations on the TRANS unit (co-executes with XDL WMMA) ----
#if __has_builtin(__builtin_amdgcn_tanhf)
__device__ __forceinline__ float tanh1(float x) { return __builtin_amdgcn_tanhf(x); }
#else
__device__ __forceinline__ float tanh1(float x) {
  float e = __builtin_amdgcn_exp2f(x * (-2.0f * LOG2E));
  return __builtin_fmaf(2.0f, __builtin_amdgcn_rcpf(1.0f + e), -1.0f);
}
#endif

// sigmoid(x + b) with hb = 0.5*b pre-folded:  0.5*tanh(0.5x + 0.5b) + 0.5
__device__ __forceinline__ float sigmoid_b(float x, float hb) {
  return __builtin_fmaf(0.5f, tanh1(__builtin_fmaf(x, 0.5f, hb)), 0.5f);
}
__device__ __forceinline__ float tanh_b(float x, float b) { return tanh1(x + b); }

__device__ __forceinline__ v8f wmma16(v16h a, v16h b, v8f c) {
  return __builtin_amdgcn_wmma_f32_16x16x32_f16(false, a, false, b, (short)0, c,
                                                false, false);
}

// B-operand fragment from LDS weight array, N-major W[n][k], row pitch `pitch`.
// lane n = n0 + lane%16; elements e<8 are K = k0 + (lane/16)*8 + e, e>=8 are K = k0 + 16 + (lane/16)*8 + (e-8).
__device__ __forceinline__ v16h ldsB(const _Float16* W, int pitch, int n0, int k0,
                                     int lane) {
  const _Float16* p = W + (n0 + (lane & 15)) * pitch + k0 + ((lane >> 4) << 3);
  v8h lo = *(const v8h*)p;
  v8h hi = *(const v8h*)(p + 16);
  return __builtin_shufflevector(lo, hi, 0, 1, 2, 3, 4, 5, 6, 7, 8, 9, 10, 11, 12,
                                 13, 14, 15);
}

// A-operand fragment from LDS H buffer, row-major [BTILE][HP] f16.
__device__ __forceinline__ v16h ldsA(const _Float16* Hbuf, int k0, int lane) {
  const _Float16* p = Hbuf + (lane & 15) * HP + k0 + ((lane >> 4) << 3);
  v8h lo = *(const v8h*)p;
  v8h hi = *(const v8h*)(p + 16);
  return __builtin_shufflevector(lo, hi, 0, 1, 2, 3, 4, 5, 6, 7, 8, 9, 10, 11, 12,
                                 13, 14, 15);
}

// A-operand fragment straight from global f32 x row (K = D = 32).
__device__ __forceinline__ v16h gloA(const float* xrow, int lane) {
  const int half = lane >> 4;
  const float* p0 = xrow + half * 8;
  const float* p1 = xrow + 16 + half * 8;
  v16h a;
#pragma unroll
  for (int e = 0; e < 8; ++e) {
    a[e]     = (_Float16)p0[e];
    a[8 + e] = (_Float16)p1[e];
  }
  return a;
}

__global__ __launch_bounds__(WAVES * 32, 1) void lstm_fused_kernel(
    const float* __restrict__ x,
    const float* __restrict__ W_ih0, const float* __restrict__ W_hh0,
    const float* __restrict__ b_ih0, const float* __restrict__ b_hh0,
    const float* __restrict__ W_ih1, const float* __restrict__ W_hh1,
    const float* __restrict__ b_ih1, const float* __restrict__ b_hh1,
    const float* __restrict__ fc1_w, const float* __restrict__ fc1_b,
    const float* __restrict__ fc2_w, const float* __restrict__ fc2_b,
    float* __restrict__ out) {
  __shared__ _Float16 sWih0[GP * D_SZ];  // 16 KB
  __shared__ _Float16 sWhh0[GP * HP];    // 32 KB
  __shared__ _Float16 sWih1[GP * HP];    // 32 KB
  __shared__ _Float16 sWhh1[GP * HP];    // 32 KB
  __shared__ float    sB0[GP];
  __shared__ float    sB1[GP];
  __shared__ _Float16 sH1[WAVES][BTILE * HP];
  __shared__ _Float16 sH2[WAVES][BTILE * HP];

  const int tid  = threadIdx.x;
  const int lane = tid & 31;
  const int wave = tid >> 5;
  const int b0   = blockIdx.x * (WAVES * BTILE) + wave * BTILE;

  // ---- cooperative weight load: pad 50->64 per gate, cast f32->f16 ----
  for (int p = tid; p < GP * D_SZ; p += blockDim.x) {
    const int n = p / D_SZ, k = p % D_SZ;
    const int g = n >> 6, j = n & 63;
    sWih0[p] = (_Float16)((j < H_SZ) ? W_ih0[(g * H_SZ + j) * D_SZ + k] : 0.0f);
  }
  for (int p = tid; p < GP * HP; p += blockDim.x) {
    const int n = p / HP, k = p % HP;
    const int g = n >> 6, j = n & 63;
    const bool ok = (j < H_SZ) && (k < H_SZ);
    const int  src = (g * H_SZ + j) * H_SZ + k;
    sWhh0[p] = (_Float16)(ok ? W_hh0[src] : 0.0f);
    sWih1[p] = (_Float16)(ok ? W_ih1[src] : 0.0f);
    sWhh1[p] = (_Float16)(ok ? W_hh1[src] : 0.0f);
  }
  for (int n = tid; n < GP; n += blockDim.x) {
    const int g = n >> 6, j = n & 63;
    sB0[n] = (j < H_SZ) ? (b_ih0[g * H_SZ + j] + b_hh0[g * H_SZ + j]) : 0.0f;
    sB1[n] = (j < H_SZ) ? (b_ih1[g * H_SZ + j] + b_hh1[g * H_SZ + j]) : 0.0f;
  }
  for (int p = lane; p < BTILE * HP; p += 32) {  // h0 = 0
    sH1[wave][p] = (_Float16)0.0f;
    sH2[wave][p] = (_Float16)0.0f;
  }
  __syncthreads();

  // Pre-scaled per-lane bias registers (sigmoid gates: 0.5*b; tanh gate: b).
  float nb0[4][4], nb1[4][4];  // [gate][jt]
#pragma unroll
  for (int g = 0; g < 4; ++g)
#pragma unroll
    for (int jt = 0; jt < 4; ++jt) {
      const float s = (g == 2) ? 1.0f : 0.5f;
      nb0[g][jt] = s * sB0[g * HP + jt * 16 + (lane & 15)];
      nb1[g][jt] = s * sB1[g * HP + jt * 16 + (lane & 15)];
    }

  float c1[4][8], c2[4][8];
#pragma unroll
  for (int jt = 0; jt < 4; ++jt)
#pragma unroll
    for (int e = 0; e < 8; ++e) { c1[jt][e] = 0.0f; c2[jt][e] = 0.0f; }

  const int    half8 = (lane >> 4) << 3;
  const float* xbase = x + ((size_t)(b0 + (lane & 15)) * T_SZ) * D_SZ;

  for (int t = 0; t < T_SZ; ++t) {
    const float* xrow = xbase + (size_t)t * D_SZ;
    if (t + 1 < T_SZ) __builtin_prefetch(xrow + D_SZ, 0, 0);  // global_prefetch
    const v16h Ax   = gloA(xrow, lane);
    const v16h Ah1a = ldsA(sH1[wave], 0, lane);   // h1_{t-1}
    const v16h Ah1b = ldsA(sH1[wave], 32, lane);

    // -------- layer 0: gates = x_t@Wih0^T + h1_{t-1}@Whh0^T (+bias in act) ----
#pragma unroll
    for (int jt = 0; jt < 4; ++jt) {
      v8f acc[4];
#pragma unroll
      for (int g = 0; g < 4; ++g) {
        const int n0 = g * HP + jt * 16;
        v8f c = {};
        c = wmma16(Ax,   ldsB(sWih0, D_SZ, n0, 0,  lane), c);
        c = wmma16(Ah1a, ldsB(sWhh0, HP,   n0, 0,  lane), c);
        c = wmma16(Ah1b, ldsB(sWhh0, HP,   n0, 32, lane), c);
        acc[g] = c;
      }
      const int j = jt * 16 + (lane & 15);
#pragma unroll
      for (int e = 0; e < 8; ++e) {
        const float ig = sigmoid_b(acc[0][e], nb0[0][jt]);
        const float fg = sigmoid_b(acc[1][e], nb0[1][jt]);
        const float gg = tanh_b(acc[2][e], nb0[2][jt]);
        const float og = sigmoid_b(acc[3][e], nb0[3][jt]);
        const float cc = __builtin_fmaf(fg, c1[jt][e], ig * gg);
        c1[jt][e] = cc;
        sH1[wave][(e + half8) * HP + j] = (_Float16)(og * tanh1(cc));
      }
    }

    // -------- layer 1: gates = h1_t@Wih1^T + h2_{t-1}@Whh1^T (+bias in act) ----
    const v16h Bh1a = ldsA(sH1[wave], 0, lane);   // h1_t (same wave, DS in-order)
    const v16h Bh1b = ldsA(sH1[wave], 32, lane);
    const v16h Ah2a = ldsA(sH2[wave], 0, lane);   // h2_{t-1}
    const v16h Ah2b = ldsA(sH2[wave], 32, lane);
#pragma unroll
    for (int jt = 0; jt < 4; ++jt) {
      v8f acc[4];
#pragma unroll
      for (int g = 0; g < 4; ++g) {
        const int n0 = g * HP + jt * 16;
        v8f c = {};
        c = wmma16(Bh1a, ldsB(sWih1, HP, n0, 0,  lane), c);
        c = wmma16(Bh1b, ldsB(sWih1, HP, n0, 32, lane), c);
        c = wmma16(Ah2a, ldsB(sWhh1, HP, n0, 0,  lane), c);
        c = wmma16(Ah2b, ldsB(sWhh1, HP, n0, 32, lane), c);
        acc[g] = c;
      }
      const int j = jt * 16 + (lane & 15);
#pragma unroll
      for (int e = 0; e < 8; ++e) {
        const float ig = sigmoid_b(acc[0][e], nb1[0][jt]);
        const float fg = sigmoid_b(acc[1][e], nb1[1][jt]);
        const float gg = tanh_b(acc[2][e], nb1[2][jt]);
        const float og = sigmoid_b(acc[3][e], nb1[3][jt]);
        const float cc = __builtin_fmaf(fg, c2[jt][e], ig * gg);
        c2[jt][e] = cc;
        sH2[wave][(e + half8) * HP + j] = (_Float16)(og * tanh1(cc));
      }
    }
  }

  // -------- FC head on h2_{T-1}: lanes 0..15, one batch row each --------
  if (lane < BTILE) {
    float hv[H_SZ];
#pragma unroll
    for (int j = 0; j < H_SZ; ++j) hv[j] = (float)sH2[wave][lane * HP + j];
    float acc2 = fc2_b[0];
    for (int ff = 0; ff < FC_SZ; ++ff) {
      float s = fc1_b[ff];
#pragma unroll
      for (int j = 0; j < H_SZ; ++j) s += hv[j] * fc1_w[ff * H_SZ + j];
      acc2 += fmaxf(s, 0.0f) * fc2_w[ff];
    }
    out[b0 + lane] = acc2;
  }
}

extern "C" void kernel_launch(void* const* d_in, const int* in_sizes, int n_in,
                              void* d_out, int out_size, void* d_ws, size_t ws_size,
                              hipStream_t stream) {
  (void)in_sizes; (void)n_in; (void)d_ws; (void)ws_size; (void)out_size;
  const float* x     = (const float*)d_in[0];
  const float* W_ih0 = (const float*)d_in[1];
  const float* W_hh0 = (const float*)d_in[2];
  const float* b_ih0 = (const float*)d_in[3];
  const float* b_hh0 = (const float*)d_in[4];
  const float* W_ih1 = (const float*)d_in[5];
  const float* W_hh1 = (const float*)d_in[6];
  const float* b_ih1 = (const float*)d_in[7];
  const float* b_hh1 = (const float*)d_in[8];
  const float* fc1_w = (const float*)d_in[9];
  const float* fc1_b = (const float*)d_in[10];
  const float* fc2_w = (const float*)d_in[11];
  const float* fc2_b = (const float*)d_in[12];
  float* out = (float*)d_out;

  dim3 grid(B_SZ / (WAVES * BTILE));  // 128 workgroups
  dim3 block(WAVES * 32);             // 2 waves
  lstm_fused_kernel<<<grid, block, 0, stream>>>(
      x, W_ih0, W_hh0, b_ih0, b_hh0, W_ih1, W_hh1, b_ih1, b_hh1,
      fc1_w, fc1_b, fc2_w, fc2_b, out);
}
